// AttentionFusion_86225763434738
// MI455X (gfx1250) — matrix operand order repaired and checked by reading, not compiled, not run on hardware
//
#include <hip/hip_runtime.h>
#include <hip/hip_bf16.h>

// ---------------------------------------------------------------------------
// Fused attention for B=8, S=2048, D=1024 on gfx1250 (MI455X).
// All GEMMs run through v_wmma_f32_16x16x32_bf16 (fp32 accumulation).
// Register-blocked: each wave computes a 32x64 output block (8 wmma / k-step).
// ---------------------------------------------------------------------------

typedef __attribute__((ext_vector_type(16))) __bf16 v16bf;
typedef __attribute__((ext_vector_type(8)))  __bf16 v8bf;
typedef __attribute__((ext_vector_type(8)))  float  v8f;

constexpr int BB    = 8;
constexpr int SS    = 2048;
constexpr int HALFD = 512;
constexpr int DD    = 1024;           // model dim
constexpr int MTOK  = BB * SS;        // 16384 tokens

// ---------------------------------------------------------------------------
// WMMA fragment loaders (wave32; k = 16*(lane/16) + s in both A and B, not the ISA table order)
// ---------------------------------------------------------------------------

// A: 16x32 bf16, row-major in memory (lda in elements).
// lanes 0-15 (row m): K = 0..15 contiguous
// lanes 16-31 (row m): K = 16..31 contiguous
__device__ __forceinline__ v16bf load_a_frag(const __bf16* __restrict__ A, int lda,
                                             int m0, int k0, int lane) {
  const int row = m0 + (lane & 15);
  const int kb  = k0 + ((lane >> 4) << 4);
  const __bf16* p = A + (size_t)row * lda + kb;
  v8bf lo = *(const v8bf*)(p);        // K = kb .. kb+7
  v8bf hi = *(const v8bf*)(p + 8);    // K = kb+8 .. kb+15
  v16bf a;
#pragma unroll
  for (int i = 0; i < 8; ++i) { a[i] = lo[i]; a[i + 8] = hi[i]; }
  return a;
}

// B: 32x16 bf16 where memory holds Wrow[N][K] row-major (B[k][n] = Wrow[n][k]).
// lane = column n; lanes 0-15 hold K 0..15, lanes 16-31 hold K 16..31;
// per lane: 16 contiguous bf16 = one 32-byte load.
__device__ __forceinline__ v16bf load_b_frag(const __bf16* __restrict__ Wrow, int ldb,
                                             int n0, int k0, int lane) {
  const int col = n0 + (lane & 15);
  const int kb  = k0 + ((lane >> 4) << 4);
  return *(const v16bf*)(Wrow + (size_t)col * ldb + kb);
}

// Register-blocked 32x64 wave tile: acc[2][4] of 16x16 fp32 tiles.
// Per k-step: 2 A-frags + 4 B-frags feed 8 wmma (A reused 4x, B reused 2x).
__device__ __forceinline__ void wmma_block_32x64(const __bf16* __restrict__ A, int lda,
                                                 const __bf16* __restrict__ Wrow, int ldb,
                                                 int m0, int n0, int kdim, int lane,
                                                 v8f acc[2][4]) {
#pragma unroll 2
  for (int k0 = 0; k0 < kdim; k0 += 32) {
    // speculative prefetch of the next weight chunk (global_prefetch_b8)
    __builtin_prefetch(Wrow + (size_t)(n0 + (lane & 15)) * ldb + k0 + 64, 0, 1);
    v16bf a0 = load_a_frag(A, lda, m0,      k0, lane);
    v16bf a1 = load_a_frag(A, lda, m0 + 16, k0, lane);
#pragma unroll
    for (int j = 0; j < 4; ++j) {
      v16bf b = load_b_frag(Wrow, ldb, n0 + 16 * j, k0, lane);
      acc[0][j] = __builtin_amdgcn_wmma_f32_16x16x32_bf16(false, a0, false, b,
                                                          (short)0, acc[0][j], false, false);
      acc[1][j] = __builtin_amdgcn_wmma_f32_16x16x32_bf16(false, a1, false, b,
                                                          (short)0, acc[1][j], false, false);
    }
  }
}

// C/D fp32 16x16 layout: lane -> col = lane&15, row = jj + 8*(lane>=16).

// ---------------------------------------------------------------------------
// Kernel 1a: concat(t,c) -> bf16 X [MTOK][DD]
// ---------------------------------------------------------------------------
__global__ void __launch_bounds__(256)
pack_x_kernel(const float* __restrict__ t_out, const float* __restrict__ c_out,
              __bf16* __restrict__ Xb) {
  size_t idx = (size_t)blockIdx.x * 256 + threadIdx.x;   // over MTOK*DD
  int d = (int)(idx & (DD - 1));
  size_t tok = idx >> 10;
  float v = (d < HALFD) ? t_out[tok * HALFD + d] : c_out[tok * HALFD + (d - HALFD)];
  Xb[idx] = (__bf16)v;
}

// ---------------------------------------------------------------------------
// Kernel 1b: fp32 weights -> bf16, three matrices selected by blockIdx.z
// ---------------------------------------------------------------------------
__global__ void __launch_bounds__(256)
pack_w_kernel(const float* __restrict__ Wq, const float* __restrict__ Wk,
              const float* __restrict__ Wv, __bf16* __restrict__ Wb3) {
  const float* src = (blockIdx.z == 0) ? Wq : (blockIdx.z == 1) ? Wk : Wv;
  size_t idx = (size_t)blockIdx.x * 256 + threadIdx.x;   // over DD*DD
  Wb3[(size_t)blockIdx.z * DD * DD + idx] = (__bf16)src[idx];
}

// ---------------------------------------------------------------------------
// Kernel 2: QKV projection.  C[m][n] = sum_k X[m][k] * W[n][k] + bias[n]
//   z==0 -> Qb [MTOK][DD] bf16     z==1 -> Kb [MTOK][DD] bf16
//   z==2 -> Vt [BB][DD][SS] bf16 (transposed store for the PV GEMM)
// block = 8 waves in 4(m) x 2(n); wave tile 32x64 -> block tile 128x128
// ---------------------------------------------------------------------------
__global__ void __launch_bounds__(256)
qkv_gemm_kernel(const __bf16* __restrict__ Xb, const __bf16* __restrict__ Wb3,
                const float* __restrict__ bq, const float* __restrict__ bk,
                const float* __restrict__ bv,
                __bf16* __restrict__ Qb, __bf16* __restrict__ Kb,
                __bf16* __restrict__ Vt) {
  const int which = blockIdx.z;
  const __bf16* Wrow = Wb3 + (size_t)which * DD * DD;
  const float*  bias = (which == 0) ? bq : (which == 1) ? bk : bv;

  const int wave = threadIdx.x >> 5;
  const int lane = threadIdx.x & 31;
  const int m0 = blockIdx.x * 128 + (wave >> 1) * 32;
  const int n0 = blockIdx.y * 128 + (wave & 1) * 64;

  v8f acc[2][4] = {};
  wmma_block_32x64(Xb, DD, Wrow, DD, m0, n0, DD, lane, acc);

  const int cn   = lane & 15;
  const int cmof = (lane >> 4) << 3;

  if (which < 2) {
    __bf16* O = (which == 0) ? Qb : Kb;
#pragma unroll
    for (int i = 0; i < 2; ++i)
#pragma unroll
      for (int j = 0; j < 4; ++j) {
        const float bn = bias[n0 + 16 * j + cn];
#pragma unroll
        for (int jj = 0; jj < 8; ++jj) {
          int row = m0 + 16 * i + cmof + jj;
          O[(size_t)row * DD + n0 + 16 * j + cn] = (__bf16)(acc[i][j][jj] + bn);
        }
      }
  } else {
    // transposed store: Vt[b][e][q]; a 128-row block never crosses a batch
    const int bidx = m0 / SS;
    __bf16* Vb = Vt + (size_t)bidx * DD * SS;
    const int qbase = m0 - bidx * SS;
#pragma unroll
    for (int i = 0; i < 2; ++i)
#pragma unroll
      for (int j = 0; j < 4; ++j) {
        const float bn = bias[n0 + 16 * j + cn];
#pragma unroll
        for (int jj = 0; jj < 8; ++jj) {
          int q = qbase + 16 * i + cmof + jj;
          Vb[(size_t)(n0 + 16 * j + cn) * SS + q] = (__bf16)(acc[i][j][jj] + bn);
        }
      }
  }
}

// ---------------------------------------------------------------------------
// Kernel 3: scores[b][q][k] = sum_d Q[b][q][d] * K[b][k][d]   (fp32 out)
// Both operands row-major over d -> same fragment loaders as above.
// ---------------------------------------------------------------------------
__global__ void __launch_bounds__(256)
scores_kernel(const __bf16* __restrict__ Qb, const __bf16* __restrict__ Kb,
              float* __restrict__ Pf) {
  const int b = blockIdx.z;
  const __bf16* A    = Qb + (size_t)b * SS * DD;
  const __bf16* Wrow = Kb + (size_t)b * SS * DD;
  float*        C    = Pf + (size_t)b * SS * SS;

  const int wave = threadIdx.x >> 5;
  const int lane = threadIdx.x & 31;
  const int m0 = blockIdx.x * 128 + (wave >> 1) * 32;
  const int n0 = blockIdx.y * 128 + (wave & 1) * 64;

  v8f acc[2][4] = {};
  wmma_block_32x64(A, DD, Wrow, DD, m0, n0, DD, lane, acc);

  const int cn   = lane & 15;
  const int cmof = (lane >> 4) << 3;
#pragma unroll
  for (int i = 0; i < 2; ++i)
#pragma unroll
    for (int j = 0; j < 4; ++j)
#pragma unroll
      for (int jj = 0; jj < 8; ++jj)
        C[(size_t)(m0 + 16 * i + cmof + jj) * SS + n0 + 16 * j + cn] = acc[i][j][jj];
}

// ---------------------------------------------------------------------------
// Kernel 4: row softmax over 2048 fp32 scores -> bf16 probabilities
// one 256-thread block per row; LDS tree reductions (max, then sum)
// ---------------------------------------------------------------------------
__global__ void __launch_bounds__(256)
softmax_kernel(const float* __restrict__ Pf, __bf16* __restrict__ Pb) {
  const size_t row = blockIdx.x;               // 0 .. BB*SS-1
  const float* in  = Pf + row * SS;
  __bf16*      out = Pb + row * SS;
  const int t = threadIdx.x;

  __shared__ float red[256];

  float lmax = -3.402823466e38f;
  for (int i = t; i < SS; i += 256) lmax = fmaxf(lmax, in[i]);
  red[t] = lmax;
  __syncthreads();
#pragma unroll
  for (int s = 128; s > 0; s >>= 1) {
    if (t < s) red[t] = fmaxf(red[t], red[t + s]);
    __syncthreads();
  }
  const float m = red[0];
  __syncthreads();

  float lsum = 0.f;
  for (int i = t; i < SS; i += 256) lsum += __expf(in[i] - m);
  red[t] = lsum;
  __syncthreads();
#pragma unroll
  for (int s = 128; s > 0; s >>= 1) {
    if (t < s) red[t] += red[t + s];
    __syncthreads();
  }
  const float inv = 1.0f / red[0];

  for (int i = t; i < SS; i += 256)
    out[i] = (__bf16)(__expf(in[i] - m) * inv);
}

// ---------------------------------------------------------------------------
// Kernel 5: out[b][q][e] = sum_k P[b][q][k] * V[b][k][e]
// Vt is [b][e][k] row-major -> contiguous B-fragments; fp32 output direct.
// ---------------------------------------------------------------------------
__global__ void __launch_bounds__(256)
out_gemm_kernel(const __bf16* __restrict__ Pb, const __bf16* __restrict__ Vt,
                float* __restrict__ Out) {
  const int b = blockIdx.z;
  const __bf16* A    = Pb + (size_t)b * SS * SS;   // [SS][SS]
  const __bf16* Wrow = Vt + (size_t)b * DD * SS;   // [DD][SS]
  float*        C    = Out + (size_t)b * SS * DD;

  const int wave = threadIdx.x >> 5;
  const int lane = threadIdx.x & 31;
  const int m0 = blockIdx.x * 128 + (wave >> 1) * 32;
  const int n0 = blockIdx.y * 128 + (wave & 1) * 64;

  v8f acc[2][4] = {};
  wmma_block_32x64(A, SS, Wrow, SS, m0, n0, SS, lane, acc);

  const int cn   = lane & 15;
  const int cmof = (lane >> 4) << 3;
#pragma unroll
  for (int i = 0; i < 2; ++i)
#pragma unroll
    for (int j = 0; j < 4; ++j)
#pragma unroll
      for (int jj = 0; jj < 8; ++jj)
        C[(size_t)(m0 + 16 * i + cmof + jj) * DD + n0 + 16 * j + cn] = acc[i][j][jj];
}

// ---------------------------------------------------------------------------
// Host-side launch.  Workspace layout (bytes):
//   Xb  bf16 [MTOK][DD]        33,554,432
//   Qb  bf16 [MTOK][DD]        33,554,432   (reused as Pb low half)
//   Kb  bf16 [MTOK][DD]        33,554,432   (reused as Pb high half)
//   Vt  bf16 [BB][DD][SS]      33,554,432
//   Wb3 bf16 [3][DD][DD]        6,291,456
//   Pf  f32  [BB][SS][SS]     134,217,728
// total ~262 MiB; Pb (bf16 probs, 64 MiB) overlays Qb+Kb after scores_kernel.
// ---------------------------------------------------------------------------
extern "C" void kernel_launch(void* const* d_in, const int* in_sizes, int n_in,
                              void* d_out, int out_size, void* d_ws, size_t ws_size,
                              hipStream_t stream) {
  (void)in_sizes; (void)n_in; (void)out_size; (void)ws_size;

  const float* t_out = (const float*)d_in[0];
  const float* c_out = (const float*)d_in[1];
  const float* Wq    = (const float*)d_in[2];
  const float* bq    = (const float*)d_in[3];
  const float* Wk    = (const float*)d_in[4];
  const float* bk    = (const float*)d_in[5];
  const float* Wv    = (const float*)d_in[6];
  const float* bv    = (const float*)d_in[7];
  float* Out = (float*)d_out;

  char* ws = (char*)d_ws;
  size_t o = 0;
  __bf16* Xb  = (__bf16*)(ws + o); o += (size_t)MTOK * DD * 2;
  __bf16* Qb  = (__bf16*)(ws + o); o += (size_t)MTOK * DD * 2;
  __bf16* Kb  = (__bf16*)(ws + o); o += (size_t)MTOK * DD * 2;
  __bf16* Vt  = (__bf16*)(ws + o); o += (size_t)MTOK * DD * 2;
  __bf16* Wb3 = (__bf16*)(ws + o); o += (size_t)3 * DD * DD * 2;
  float*  Pf  = (float*) (ws + o); o += (size_t)BB * SS * SS * 4;
  __bf16* Pb  = Qb;   // 64 MiB overlay of Qb+Kb, dead after scores_kernel

  pack_x_kernel<<<dim3((MTOK * DD) / 256), dim3(256), 0, stream>>>(t_out, c_out, Xb);
  pack_w_kernel<<<dim3((DD * DD) / 256, 1, 3), dim3(256), 0, stream>>>(Wq, Wk, Wv, Wb3);

  qkv_gemm_kernel<<<dim3(MTOK / 128, DD / 128, 3), dim3(256), 0, stream>>>(
      Xb, Wb3, bq, bk, bv, Qb, Kb, Vt);

  scores_kernel<<<dim3(SS / 128, SS / 128, BB), dim3(256), 0, stream>>>(Qb, Kb, Pf);

  softmax_kernel<<<dim3(BB * SS), dim3(256), 0, stream>>>(Pf, Pb);

  out_gemm_kernel<<<dim3(SS / 128, DD / 128, BB), dim3(256), 0, stream>>>(Pb, Vt, Out);
}
